// Sub2GeneDifferCrossAttn_12154757448325
// MI455X (gfx1250) — matrix operand order repaired and checked by reading, not compiled, not run on hardware
//
#include <hip/hip_runtime.h>
#include <hip/hip_bf16.h>
#include <math.h>

// ---- problem constants (from reference) ----
#define BB   8
#define SS   512
#define LL   4096        // P*G
#define EE   768
#define GEd  32
#define HDm  384         // E/2
#define KV_N (2*EE)      // 1536

static constexpr float LAMBDA_INIT_C = 0.2f;                  // 0.8 - 0.6*exp(0)
static constexpr float SCALING_C     = 0.051031036307982884f; // 384^-0.5

typedef __attribute__((ext_vector_type(16))) __bf16 v16bf;
typedef __attribute__((ext_vector_type(8)))  __bf16 v8bf;
typedef __attribute__((ext_vector_type(8)))  float  v8f;
typedef __attribute__((ext_vector_type(2)))  float  v2f;

// 16-byte vector load of 8 bf16
__device__ inline v8bf ld8(const __bf16* p) { return *(const v8bf*)p; }
// pack two v8bf into one v16bf (pure register packing, no VALU converts)
__device__ inline v16bf pack16(v8bf lo, v8bf hi) {
    return __builtin_shufflevector(lo, hi, 0, 1, 2, 3, 4, 5, 6, 7,
                                           8, 9, 10, 11, 12, 13, 14, 15);
}

// =====================================================================
// f32 WMMA 16x16x4 core (exact). One wave per 16x16 C tile.
// A frag: lane<16: M=lane, K=k..k+1 ; lane>=16: M=lane-16, K=k+2..k+3
// B frag: col N=lane&15 ; K = k + 2*(lane>>4) + {0,1}
// C/D:    VGPR v -> row v + 8*(lane>=16), col lane&15
// =====================================================================
__device__ inline v8f gemm_core_f32(const float* __restrict__ A,
                                    const float* __restrict__ B,
                                    int K, int lda, int ldb, int lane)
{
    const int half = lane >> 4;
    const int idx  = lane & 15;
    v8f acc = {};
    const float* Ar = A + (size_t)idx * lda + half * 2;
    const float* Br = B + (size_t)(half * 2) * ldb + idx;
    for (int k = 0; k < K; k += 4) {
        v2f a, b;
        a.x = Ar[k];
        a.y = Ar[k + 1];
        b.x = Br[(size_t)k * ldb];
        b.y = Br[(size_t)(k + 1) * ldb];
        acc = __builtin_amdgcn_wmma_f32_16x16x4_f32(
            false, a, false, b, (short)0, acc, false, false);
    }
    return acc;
}

// ---- final projection: plain f32 output -----------------------------
__global__ __launch_bounds__(32) void gemm_wmma_f32(
    const float* __restrict__ A, const float* __restrict__ B,
    float* __restrict__ C, int K, int lda, int ldb, int ldc)
{
    const int m0 = blockIdx.x * 16, n0 = blockIdx.y * 16;
    const int lane = threadIdx.x, half = lane >> 4, idx = lane & 15;
    v8f acc = gemm_core_f32(A + (size_t)m0 * lda, B + n0, K, lda, ldb, lane);
    float* Cr = C + (size_t)(m0 + half * 8) * ldc + n0 + idx;
#pragma unroll
    for (int v = 0; v < 8; ++v) Cr[(size_t)v * ldc] = acc[v];
}

// ---- Q projection: epilogue emits bf16 with SCALING folded ----------
__global__ __launch_bounds__(32) void gemm_q_bf16(
    const float* __restrict__ A, const float* __restrict__ B,
    __bf16* __restrict__ Qbf, int K, int lda, int ldb)
{
    const int m0 = blockIdx.x * 16, n0 = blockIdx.y * 16;
    const int lane = threadIdx.x, half = lane >> 4, idx = lane & 15;
    v8f acc = gemm_core_f32(A + (size_t)m0 * lda, B + n0, K, lda, ldb, lane);
    __bf16* Cr = Qbf + (size_t)(m0 + half * 8) * EE + n0 + idx;
#pragma unroll
    for (int v = 0; v < 8; ++v) Cr[(size_t)v * EE] = (__bf16)(acc[v] * SCALING_C);
}

// ---- KV projection: K half -> row-major bf16 ; V half -> transposed --
__global__ __launch_bounds__(32) void gemm_kv_bf16(
    const float* __restrict__ A, const float* __restrict__ B,
    __bf16* __restrict__ Kbf,   // (B, L, E) row-major
    __bf16* __restrict__ Vt)    // (B, E, L) transposed
{
    const int m0 = blockIdx.x * 16, n0 = blockIdx.y * 16;
    const int lane = threadIdx.x, half = lane >> 4, idx = lane & 15;
    v8f acc = gemm_core_f32(A + (size_t)m0 * GEd, B + n0, GEd, GEd, KV_N, lane);
    const int n = n0 + idx;
    if (n < EE) {
        __bf16* Cr = Kbf + (size_t)(m0 + half * 8) * EE + n;
#pragma unroll
        for (int v = 0; v < 8; ++v) Cr[(size_t)v * EE] = (__bf16)acc[v];
    } else {
#pragma unroll
        for (int v = 0; v < 8; ++v) {
            const int row = m0 + half * 8 + v;        // global key row: b*L + l
            const int bb  = row >> 12;                // / L
            const int lr  = row & (LL - 1);
            Vt[((size_t)bb * EE + (n - EE)) * LL + lr] = (__bf16)acc[v];
        }
    }
}

// =====================================================================
// bf16 score tile: 16x16 = A(16xKD) @ B(16xKD)^T, both row-major bf16,
// contiguous-K. Inner loop: 4x global_load_b128 + 1x wmma per k-step.
// A frag segs at +0/+16 elems; B frag segs at +0/+16 elems (ISA 7.12.2).
// =====================================================================
__device__ inline v8f tile_bf16(const __bf16* __restrict__ Arow, int a_ld,
                                const __bf16* __restrict__ Brow, int b_ld,
                                int KD, int lane)
{
    const int half = lane >> 4;
    const int idx  = lane & 15;
    v8f acc = {};
    const __bf16* Ap0 = Arow + (size_t)idx * a_ld + half * 8;
    const __bf16* Bp0 = Brow + (size_t)idx * b_ld + half * 8;
    for (int k = 0; k < KD; k += 32) {
        v16bf a = pack16(ld8(Ap0 + k), ld8(Ap0 + k + 16));
        v16bf b = pack16(ld8(Bp0 + k), ld8(Bp0 + k + 16));
        acc = __builtin_amdgcn_wmma_f32_16x16x32_bf16(
            false, a, false, b, (short)0, acc, false, false);
    }
    return acc;
}

// =====================================================================
// Pass 1: online softmax stats (max m, sum d) per (b, head, s)
// grid = (B, S/16), block = 32. Lane -> (head = lane>>4, row = lane&15)
// =====================================================================
__global__ __launch_bounds__(32) void attn_pass1(
    const __bf16* __restrict__ Qbf, const __bf16* __restrict__ Kbf,
    const int* __restrict__ qmask, const int* __restrict__ kmask,
    float* __restrict__ mOut, float* __restrict__ dOut)
{
    __shared__ float tile[2][16][16];
    const int b = blockIdx.x, s0 = blockIdx.y * 16;
    const int lane = threadIdx.x, half = lane >> 4, idx = lane & 15;
    const int h = half, r = idx;

    const __bf16* Qb = Qbf + ((size_t)b * SS + s0) * EE;
    const __bf16* Kb = Kbf + (size_t)b * LL * EE;

    const int qm = qmask[b * SS + s0 + r];
    float m = -3.0e38f, d = 0.0f;

    for (int lt = 0; lt < LL / 16; ++lt) {
        const int l0 = lt * 16;
        const __bf16* Kt = Kb + (size_t)l0 * EE;
        v8f t0 = tile_bf16(Qb,       EE, Kt,       EE, HDm, lane);
        v8f t1 = tile_bf16(Qb + HDm, EE, Kt + HDm, EE, HDm, lane);
#pragma unroll
        for (int v = 0; v < 8; ++v) {
            tile[0][v + half * 8][idx] = t0[v];
            tile[1][v + half * 8][idx] = t1[v];
        }
        __syncthreads();
#pragma unroll
        for (int c = 0; c < 16; ++c) {
            const int km = kmask[b * LL + l0 + c];
            float sc = (qm && km) ? tile[h][r][c] : -1.0e20f;
            if (sc > m) { d *= __expf(m - sc); m = sc; }
            d += __expf(sc - m);
        }
        __syncthreads();
    }
    mOut[((size_t)b * 2 + h) * SS + s0 + r] = m;
    dOut[((size_t)b * 2 + h) * SS + s0 + r] = d;
}

// =====================================================================
// Pass 2: diff = p0/d0 - lam*p1/d1 (raw) -> d_out diff region; row min.
// Lane covers (row = lane&15, cols 8*(lane>>4)..+7); pair-combine via
// __shfl_xor(.,16) (wave32).
// =====================================================================
__global__ __launch_bounds__(32) void attn_pass2(
    const __bf16* __restrict__ Qbf, const __bf16* __restrict__ Kbf,
    const int* __restrict__ qmask, const int* __restrict__ kmask,
    const float* __restrict__ mS, const float* __restrict__ dS,
    const float* __restrict__ lamPtr,
    float* __restrict__ diffOut, float* __restrict__ rowMin)
{
    __shared__ float tile[2][16][16];
    const int b = blockIdx.x, s0 = blockIdx.y * 16;
    const int lane = threadIdx.x, half = lane >> 4, idx = lane & 15;
    const int r = idx;

    const float lam = lamPtr[0];
    const __bf16* Qb = Qbf + ((size_t)b * SS + s0) * EE;
    const __bf16* Kb = Kbf + (size_t)b * LL * EE;

    const float m0 = mS[((size_t)b * 2 + 0) * SS + s0 + r];
    const float d0 = dS[((size_t)b * 2 + 0) * SS + s0 + r] + 1e-8f;
    const float m1 = mS[((size_t)b * 2 + 1) * SS + s0 + r];
    const float d1 = dS[((size_t)b * 2 + 1) * SS + s0 + r] + 1e-8f;
    const int   qm = qmask[b * SS + s0 + r];

    float mn = 3.0e38f;
    float* drow = diffOut + ((size_t)(b * SS + s0 + r)) * LL;

    for (int lt = 0; lt < LL / 16; ++lt) {
        const int l0 = lt * 16;
        const __bf16* Kt = Kb + (size_t)l0 * EE;
        v8f t0 = tile_bf16(Qb,       EE, Kt,       EE, HDm, lane);
        v8f t1 = tile_bf16(Qb + HDm, EE, Kt + HDm, EE, HDm, lane);
#pragma unroll
        for (int v = 0; v < 8; ++v) {
            tile[0][v + half * 8][idx] = t0[v];
            tile[1][v + half * 8][idx] = t1[v];
        }
        __syncthreads();
#pragma unroll
        for (int c8 = 0; c8 < 8; ++c8) {
            const int c  = half * 8 + c8;
            const int km = kmask[b * LL + l0 + c];
            const bool ok = qm && km;
            float sc0 = ok ? tile[0][r][c] : -1.0e20f;
            float sc1 = ok ? tile[1][r][c] : -1.0e20f;
            float p0 = __expf(sc0 - m0) / d0;
            float p1 = __expf(sc1 - m1) / d1;
            float dv = p0 - lam * p1;
            drow[l0 + c] = dv;
            mn = fminf(mn, dv);
        }
        __syncthreads();
    }
    mn = fminf(mn, __shfl_xor(mn, 16, 32));
    if (half == 0) rowMin[b * SS + s0 + r] = mn;
}

// =====================================================================
// diff fixup: diff = (diff - rowmin + 1e-20) masked; dual-write f32
// (final output) + bf16 (PV GEMM operand)
// =====================================================================
__global__ __launch_bounds__(256) void diff_fixup(
    float* __restrict__ diff, __bf16* __restrict__ diffBf,
    const float* __restrict__ rowMin,
    const int* __restrict__ qmask, const int* __restrict__ kmask)
{
    size_t i = (size_t)blockIdx.x * 256 + threadIdx.x;
    const int    l   = (int)(i & (LL - 1));
    const size_t row = i >> 12;            // / LL
    const int    s   = (int)(row & (SS - 1));
    const int    b   = (int)(row >> 9);    // / SS
    float v = diff[i] - rowMin[row] + 1e-20f;
    const bool ok = qmask[b * SS + s] && kmask[b * LL + l];
    v = ok ? v : 0.0f;
    diff[i]   = v;
    diffBf[i] = (__bf16)v;
}

// =====================================================================
// PV GEMM (bf16 WMMA): attn_out = diff(BS x L) @ V(b; L x E)
// A = diffBf contiguous-K; B = Vt (E x L transposed) -> contiguous-K too.
// =====================================================================
__global__ __launch_bounds__(32) void pv_gemm(
    const __bf16* __restrict__ diffBf, const __bf16* __restrict__ Vt,
    float* __restrict__ Out)
{
    const int m0 = blockIdx.x * 16, n0 = blockIdx.y * 16;
    const int lane = threadIdx.x, half = lane >> 4, idx = lane & 15;
    const int bb = m0 / SS;

    const __bf16* Ap0 = diffBf + ((size_t)m0 + idx) * LL + half * 8;
    const __bf16* Bp0 = Vt + ((size_t)bb * EE + n0 + idx) * LL + half * 8;

    v8f acc = {};
    for (int k = 0; k < LL; k += 32) {
        v16bf a = pack16(ld8(Ap0 + k), ld8(Ap0 + k + 16));
        v16bf b = pack16(ld8(Bp0 + k), ld8(Bp0 + k + 16));
        acc = __builtin_amdgcn_wmma_f32_16x16x32_bf16(
            false, a, false, b, (short)0, acc, false, false);
    }
    float* Cp = Out + (size_t)(m0 + half * 8) * EE + n0 + idx;
#pragma unroll
    for (int v = 0; v < 8; ++v) Cp[(size_t)v * EE] = acc[v];
}

// =====================================================================
// RMS norm (in place): x *= rsqrt(mean(x^2)+1e-5) * ln_w * (1-LAMBDA_INIT)
// =====================================================================
__global__ __launch_bounds__(256) void rmsnorm(
    float* __restrict__ X, const float* __restrict__ lnw)
{
    __shared__ float red[8];
    float* x = X + (size_t)blockIdx.x * EE;
    float ss = 0.0f;
    for (int e = threadIdx.x; e < EE; e += 256) { float v = x[e]; ss += v * v; }
#pragma unroll
    for (int o = 16; o > 0; o >>= 1) ss += __shfl_xor(ss, o, 32);
    if ((threadIdx.x & 31) == 0) red[threadIdx.x >> 5] = ss;
    __syncthreads();
    if (threadIdx.x == 0) {
        float t = 0.0f;
#pragma unroll
        for (int i = 0; i < 8; ++i) t += red[i];
        red[0] = t / (float)EE;
    }
    __syncthreads();
    const float scale = rsqrtf(red[0] + 1e-5f) * (1.0f - LAMBDA_INIT_C);
    for (int e = threadIdx.x; e < EE; e += 256) x[e] = x[e] * scale * lnw[e];
}

// =====================================================================
// lambda scalar
// =====================================================================
__global__ __launch_bounds__(128) void lam_kernel(
    const float* __restrict__ lq1, const float* __restrict__ lk1,
    const float* __restrict__ lq2, const float* __restrict__ lk2,
    float* __restrict__ lamOut)
{
    __shared__ float r1[4], r2[4];
    float s1 = 0.0f, s2 = 0.0f;
    for (int i = threadIdx.x; i < HDm; i += 128) {
        s1 += lq1[i] * lk1[i];
        s2 += lq2[i] * lk2[i];
    }
#pragma unroll
    for (int o = 16; o > 0; o >>= 1) {
        s1 += __shfl_xor(s1, o, 32);
        s2 += __shfl_xor(s2, o, 32);
    }
    if ((threadIdx.x & 31) == 0) { r1[threadIdx.x >> 5] = s1; r2[threadIdx.x >> 5] = s2; }
    __syncthreads();
    if (threadIdx.x == 0) {
        float a = 0.0f, b = 0.0f;
#pragma unroll
        for (int i = 0; i < 4; ++i) { a += r1[i]; b += r2[i]; }
        lamOut[0] = __expf(a) - __expf(b) + LAMBDA_INIT_C;
    }
}

// =====================================================================
extern "C" void kernel_launch(void* const* d_in, const int* in_sizes, int n_in,
                              void* d_out, int out_size, void* d_ws, size_t ws_size,
                              hipStream_t stream)
{
    const float* query  = (const float*)d_in[0];   // B,S,E
    const float* key    = (const float*)d_in[1];   // B,P,G,GE -> (B*L) x GE
    const int*   qmask  = (const int*)  d_in[2];   // B,S
    const int*   kmask  = (const int*)  d_in[3];   // B,P,G -> B*L
    const float* Wq     = (const float*)d_in[4];   // E,E
    const float* Wkv    = (const float*)d_in[5];   // GE,2E
    const float* Wout   = (const float*)d_in[6];   // E,E
    const float* lq1    = (const float*)d_in[7];
    const float* lk1    = (const float*)d_in[8];
    const float* lq2    = (const float*)d_in[9];
    const float* lk2    = (const float*)d_in[10];
    const float* lnw    = (const float*)d_in[11];

    float* outPtr  = (float*)d_out;                       // B*S*E
    float* diffPtr = outPtr + (size_t)BB * SS * EE;       // B*S*L

    // workspace carve-up (16B-aligned segments)
    __bf16* Qbf  = (__bf16*)d_ws;                         // 4096*768
    __bf16* Kbf  = Qbf  + (size_t)BB * SS * EE;           // 32768*768
    __bf16* Vt   = Kbf  + (size_t)BB * LL * EE;           // 8*768*4096
    __bf16* dBf  = Vt   + (size_t)BB * EE * LL;           // 4096*4096
    float*  AO   = (float*)(dBf + (size_t)BB * SS * LL);  // B*S*E f32
    float*  mS   = AO   + (size_t)BB * SS * EE;           // B*2*S
    float*  dS   = mS   + (size_t)BB * 2 * SS;
    float*  rowMin = dS + (size_t)BB * 2 * SS;            // B*S
    float*  lam  = rowMin + (size_t)BB * SS;              // 1 (+pad)

    // 0) lambda scalar
    lam_kernel<<<1, 128, 0, stream>>>(lq1, lk1, lq2, lk2, lam);

    // 1) Qbf = bf16( (query @ Wq) * SCALING )   (M=4096, N=768, K=768)
    gemm_q_bf16<<<dim3((BB * SS) / 16, EE / 16), 32, 0, stream>>>(
        query, Wq, Qbf, EE, EE, EE);

    // 2) KV = key @ Wkv -> Kbf (row-major) + Vt (transposed) bf16
    gemm_kv_bf16<<<dim3((BB * LL) / 16, KV_N / 16), 32, 0, stream>>>(
        key, Wkv, Kbf, Vt);

    // 3) softmax stats per (b, head, s)
    attn_pass1<<<dim3(BB, SS / 16), 32, 0, stream>>>(
        Qbf, Kbf, qmask, kmask, mS, dS);

    // 4) raw diff + per-row min
    attn_pass2<<<dim3(BB, SS / 16), 32, 0, stream>>>(
        Qbf, Kbf, qmask, kmask, mS, dS, lam, diffPtr, rowMin);

    // 5) diff fixup (min-subtract + mask), dual f32/bf16 write
    diff_fixup<<<(BB * SS * LL) / 256, 256, 0, stream>>>(
        diffPtr, dBf, rowMin, qmask, kmask);

    // 6) attn_out = diff @ V   (M=4096, N=768, K=4096, bf16 WMMA)
    pv_gemm<<<dim3((BB * SS) / 16, EE / 16), 32, 0, stream>>>(dBf, Vt, AO);

    // 7) RMS norm in place (+ ln_weight, (1-lambda_init) scale)
    rmsnorm<<<BB * SS, 256, 0, stream>>>(AO, lnw);

    // 8) out = attn_out @ Wout  (M=4096, N=768, K=768, f32 WMMA)
    gemm_wmma_f32<<<dim3((BB * SS) / 16, EE / 16), 32, 0, stream>>>(
        AO, Wout, outPtr, EE, EE, EE, EE);
}
